// CODAPromptPool_55963423866981
// MI455X (gfx1250) — compile-verified, matrix operand order checked by reading, not run on hardware
//
#include <hip/hip_runtime.h>
#include <hip/hip_bf16.h>
#include <math.h>

// Problem constants (match reference)
#define BB 8192
#define PP 100
#define LL 8
#define DD 768
#define KK 5
#define HH 4
#define HD 192          // DD / HH
#define PL (PP * LL)    // 800 distinct value rows

typedef _Float16 half8  __attribute__((ext_vector_type(8)));
typedef _Float16 half16 __attribute__((ext_vector_type(16)));
typedef float    f32x8  __attribute__((ext_vector_type(8)));
typedef unsigned int u32x4 __attribute__((ext_vector_type(4)));
typedef int      i32x4 __attribute__((ext_vector_type(4)));
typedef int      i32x8 __attribute__((ext_vector_type(8)));

// ---------------------------------------------------------------------------
// TDM descriptor helpers (CDNA5 ISA ch.8: D# groups).
// Group 0: count=1 | lds_addr | global_addr[56:0] | type=2
// ---------------------------------------------------------------------------
static __device__ __forceinline__ u32x4 tdm_group0(unsigned lds_addr,
                                                   unsigned long long gaddr) {
    u32x4 g;
    g[0] = 1u;                                   // count=1 (valid), user mode
    g[1] = lds_addr;                             // LDS byte address
    g[2] = (unsigned)(gaddr & 0xFFFFFFFFu);      // global_addr[31:0]
    g[3] = (unsigned)((gaddr >> 32) & 0x01FFFFFFu) | (2u << 30); // [56:32] | type=2
    return g;
}

// Group 1: data_size=2B, pad_enable, pad_interval=16DW (64B), pad_amount=4DW
// (16B) -> LDS row stride 80B = 40 halves, matching LDSS below.
static __device__ __forceinline__ i32x8 tdm_group1(unsigned tdim0, unsigned tdim1,
                                                   unsigned tile0, unsigned tile1,
                                                   unsigned long long stride0) {
    i32x8 g;
    g[0] = (1 << 16)      // data_size = 2 bytes
         | (1 << 20)      // pad_enable
         | (3 << 22)      // pad_interval: 16 DWORDs
         | (3 << 25);     // pad_amount:   4 DWORDs
    g[1] = (int)((tdim0 & 0xFFFFu) << 16);                       // tensor_dim0 lo
    g[2] = (int)(((tdim0 >> 16) & 0xFFFFu) | ((tdim1 & 0xFFFFu) << 16));
    g[3] = (int)(((tdim1 >> 16) & 0xFFFFu) | (tile0 << 16));     // tile_dim0
    g[4] = (int)(tile1 & 0xFFFFu);                               // tile_dim1, tile_dim2=0
    g[5] = (int)(stride0 & 0xFFFFFFFFull);                       // tensor_dim0_stride lo
    g[6] = (int)((stride0 >> 32) & 0xFFFFull);                   // stride hi, stride1=0
    g[7] = 0;
    return g;
}

#if defined(__clang_major__) && __clang_major__ >= 23
#define TDM_LOAD(g0, g1, g2, g3)                                                \
    __builtin_amdgcn_tensor_load_to_lds((g0), (g1), (g2), (g3),                 \
                                        (i32x8){0, 0, 0, 0, 0, 0, 0, 0}, 0)
#else
#define TDM_LOAD(g0, g1, g2, g3)                                                \
    __builtin_amdgcn_tensor_load_to_lds((g0), (g1), (g2), (g3), 0)
#endif

// ---------------------------------------------------------------------------
// f32 -> f16 conversion (grid-stride)
// ---------------------------------------------------------------------------
__global__ void cvt_f32_to_f16(const float* __restrict__ src,
                               _Float16* __restrict__ dst, int n) {
    for (int i = blockIdx.x * blockDim.x + threadIdx.x; i < n;
         i += gridDim.x * blockDim.x)
        dst[i] = (_Float16)src[i];
}

// ---------------------------------------------------------------------------
// Normalize keys, store TRANSPOSED: knT[d * PP + p]
// ---------------------------------------------------------------------------
__global__ __launch_bounds__(128)
void keynorm_kernel(const float* __restrict__ keys, float* __restrict__ knT) {
    __shared__ float red[128];
    const int p = blockIdx.x;
    const int t = threadIdx.x;
    float ss = 0.f;
    for (int i = t; i < DD; i += 128) {
        float v = keys[p * DD + i];
        ss += v * v;
    }
    red[t] = ss;
    __syncthreads();
    for (int s = 64; s > 0; s >>= 1) {
        if (t < s) red[t] += red[t + s];
        __syncthreads();
    }
    float inv = 1.0f / fmaxf(sqrtf(red[0]), 1e-12f);
    for (int i = t; i < DD; i += 128)
        knT[i * PP + p] = keys[p * DD + i] * inv;
}

// ---------------------------------------------------------------------------
// WMMA GEMM with TDM-staged, double-buffered LDS tiles.
//   C[M x N] = A[M x Kd](f16,rm) * W[N x Kd]^T(f16,rm) + bias[N]  (C f32)
// 256 threads = 8 waves; block tile 128x128; wave tile 32x64.
// TDM: wave 0 issues one 128x32-half 2D tile DMA per operand per K-step,
// padded to a 40-half LDS row stride; A-side M overrun zero-filled via
// tensor_dim1. Next step's DMA overlaps current step's WMMAs.
// The asm-escape barrier makes the TDM's invisible LDS writes "visible" to
// LLVM so the ds_load fragment reads are not folded away.
// NOTE: assumes smem[] sits at LDS offset 0 (sole static allocation).
// ---------------------------------------------------------------------------
#define LDSS 40   // padded LDS row stride in halves (80B)

__global__ __launch_bounds__(256)
void gemm_f16_wmma(const _Float16* __restrict__ A, const _Float16* __restrict__ W,
                   const float* __restrict__ bias, float* __restrict__ C,
                   int M, int N, int Kd) {
    __shared__ _Float16 smem[4 * 128 * LDSS];      // [buf][A|B][128][LDSS]
    const unsigned PAIR_HALVES = 2u * 128u * LDSS; // halves per buffer pair
    const unsigned PAIR_BYTES  = PAIR_HALVES * 2u; // 20480
    const unsigned BOFF_BYTES  = 128u * LDSS * 2u; // 10240 (B tile within pair)

    const int tid  = threadIdx.x;
    const int lane = tid & 31;
    const int wave = tid >> 5;
    const int wm   = wave & 3;   // 4 waves tile M: wm*32
    const int wn   = wave >> 2;  // 2 waves tile N: wn*64
    const int blockM = blockIdx.x * 128;
    const int blockN = blockIdx.y * 128;

    unsigned arows = (unsigned)(M - blockM);
    if (arows > 128u) arows = 128u;
    const unsigned long long abase =
        (unsigned long long)A + (unsigned long long)blockM * Kd * 2ull;
    const unsigned long long bbase =
        (unsigned long long)W + (unsigned long long)blockN * Kd * 2ull;

    const i32x8 g1a = tdm_group1((unsigned)Kd, arows, 32u, 128u,
                                 (unsigned long long)Kd);
    const i32x8 g1b = tdm_group1((unsigned)Kd, 128u, 32u, 128u,
                                 (unsigned long long)Kd);
    const i32x4 gz = {0, 0, 0, 0};

    // prologue: stage K-step 0 into buffer 0 (TDM ignores EXEC; only wave 0
    // reaches the instructions via the execz-branch on tid<32)
    if (tid < 32) {
        TDM_LOAD(tdm_group0(0u, abase), g1a, gz, gz);
        TDM_LOAD(tdm_group0(BOFF_BYTES, bbase), g1b, gz, gz);
    }

    f32x8 acc[2][4];
    for (int ni = 0; ni < 4; ++ni) {
        const int ng = blockN + wn * 64 + ni * 16 + (lane & 15);
        const float bv = bias ? bias[ng] : 0.f;
        for (int mi = 0; mi < 2; ++mi)
            for (int j = 0; j < 8; ++j) acc[mi][ni][j] = bv;
    }

    for (int k0 = 0; k0 < Kd; k0 += 32) {
        const int buf = (k0 >> 5) & 1;
        // Wave-scalar wait; no-op for waves with TENSORcnt==0, so run it
        // unconditionally (avoids exec save/restore around an SOPP).
        __builtin_amdgcn_s_wait_tensorcnt(0);
        __syncthreads();  // cur buffer ready; prev buffer fully consumed

        // Compiler barrier: TDM wrote smem[] invisibly; force LLVM to treat
        // smem as modified so the fragment ds_loads below are materialized
        // here (and not folded to undef or hoisted above the wait).
        asm volatile("" : : "s"(&smem[0]) : "memory");

        const int nk = k0 + 32;
        if (nk < Kd && tid < 32) {   // overlap next DMA with current WMMAs
            const unsigned nb = (unsigned)((buf ^ 1)) * PAIR_BYTES;
            TDM_LOAD(tdm_group0(nb, abase + (unsigned long long)nk * 2ull),
                     g1a, gz, gz);
            TDM_LOAD(tdm_group0(nb + BOFF_BYTES,
                                bbase + (unsigned long long)nk * 2ull),
                     g1b, gz, gz);
        }

        const _Float16* Ash = smem + (size_t)buf * PAIR_HALVES;
        const _Float16* Bsh = Ash + 128 * LDSS;

        const int r    = lane & 15;
        const int ksub = (lane >> 4) * 8;
        half16 afrag[2], bfrag[4];
        for (int mi = 0; mi < 2; ++mi) {
            const int row = wm * 32 + mi * 16 + r;
            half8 lo = *(const half8*)(&Ash[row * LDSS + ksub]);
            half8 hi = *(const half8*)(&Ash[row * LDSS + ksub + 16]);
            afrag[mi] = __builtin_shufflevector(lo, hi, 0,1,2,3,4,5,6,7,
                                                        8,9,10,11,12,13,14,15);
        }
        for (int ni = 0; ni < 4; ++ni) {
            const int row = wn * 64 + ni * 16 + r;
            half8 lo = *(const half8*)(&Bsh[row * LDSS + ksub]);
            half8 hi = *(const half8*)(&Bsh[row * LDSS + ksub + 16]);
            bfrag[ni] = __builtin_shufflevector(lo, hi, 0,1,2,3,4,5,6,7,
                                                        8,9,10,11,12,13,14,15);
        }
        for (int mi = 0; mi < 2; ++mi)
            for (int ni = 0; ni < 4; ++ni)
                acc[mi][ni] = __builtin_amdgcn_wmma_f32_16x16x32_f16(
                    false, afrag[mi], false, bfrag[ni],
                    (short)0, acc[mi][ni], false, false);
    }

    const int r  = lane & 15;
    const int mo = (lane >> 4) * 8;
    for (int mi = 0; mi < 2; ++mi)
        for (int ni = 0; ni < 4; ++ni) {
            const int ng = blockN + wn * 64 + ni * 16 + r;
            for (int j = 0; j < 8; ++j) {
                const int mg = blockM + wm * 32 + mi * 16 + mo + j;
                if (mg < M) C[(size_t)mg * N + ng] = acc[mi][ni][j];
            }
        }
}

// ---------------------------------------------------------------------------
// Per-row similarity + top-5 (scale-invariant: x need not be normalized).
// ---------------------------------------------------------------------------
__global__ __launch_bounds__(128)
void topk_kernel(const float* __restrict__ x, const float* __restrict__ knT,
                 int* __restrict__ topidx) {
    __shared__ float xs[DD];
    __shared__ float sims[128];
    __shared__ float rv[128];
    __shared__ int   ri[128];
    const int b = blockIdx.x;
    const int t = threadIdx.x;

    for (int i = t; i < DD; i += 128) xs[i] = x[(size_t)b * DD + i];
    __syncthreads();

    float s = -INFINITY;
    if (t < PP) {
        float acc = 0.f;
        for (int i = 0; i < DD; ++i) acc += xs[i] * knT[i * PP + t];
        s = acc;
    }
    sims[t] = s;
    __syncthreads();

    for (int k = 0; k < KK; ++k) {
        rv[t] = sims[t];
        ri[t] = t;
        __syncthreads();
        for (int stp = 64; stp > 0; stp >>= 1) {
            if (t < stp) {
                const float v2 = rv[t + stp];
                const int   i2 = ri[t + stp];
                if (v2 > rv[t] || (v2 == rv[t] && i2 < ri[t])) {
                    rv[t] = v2; ri[t] = i2;
                }
            }
            __syncthreads();
        }
        if (t == 0) {
            topidx[b * KK + k] = ri[0];
            sims[ri[0]] = -INFINITY;
        }
        __syncthreads();
    }
}

// ---------------------------------------------------------------------------
// Attention over 40 gathered (pre-projected) kv rows; writes ctx as f16.
// ---------------------------------------------------------------------------
__global__ __launch_bounds__(256)
void attn_kernel(const float* __restrict__ q, const float* __restrict__ kproj,
                 const float* __restrict__ vproj, const int* __restrict__ topidx,
                 _Float16* __restrict__ ctx_h) {
    __shared__ float qs[DD];
    __shared__ int   fr[KK * LL];
    __shared__ float sc[HH * KK * LL];
    __shared__ float mx[HH];
    __shared__ float dn[HH];
    const int b = blockIdx.x;
    const int t = threadIdx.x;
    const int NS = KK * LL;  // 40

    for (int i = t; i < DD; i += 256) qs[i] = q[(size_t)b * DD + i];
    if (t < NS) fr[t] = topidx[b * KK + (t >> 3)] * LL + (t & 7);
    __syncthreads();

    if (t < HH * NS) {
        const int h = t / NS, s = t - h * NS;
        const float* kr = kproj + (size_t)fr[s] * DD + h * HD;
        float acc = 0.f;
        for (int i = 0; i < HD; ++i) acc += qs[h * HD + i] * kr[i];
        sc[t] = acc * 0.072168783648703332f;  // 1/sqrt(192)
    }
    __syncthreads();
    if (t < HH) {
        float m = -INFINITY;
        for (int s = 0; s < NS; ++s) m = fmaxf(m, sc[t * NS + s]);
        mx[t] = m;
    }
    __syncthreads();
    if (t < HH * NS) sc[t] = __expf(sc[t] - mx[t / NS]);
    __syncthreads();
    if (t < HH) {
        float d = 0.f;
        for (int s = 0; s < NS; ++s) d += sc[t * NS + s];
        dn[t] = d;
    }
    __syncthreads();
    if (t < HH * NS) sc[t] /= dn[t / NS];
    __syncthreads();

    for (int o = t; o < DD; o += 256) {
        const int h = o / HD, d = o - h * HD;
        float acc = 0.f;
        for (int s = 0; s < NS; ++s)
            acc += sc[h * NS + s] * vproj[(size_t)fr[s] * DD + h * HD + d];
        ctx_h[(size_t)b * DD + o] = (_Float16)acc;
    }
}

// ---------------------------------------------------------------------------
// Residual + LayerNorm.
// ---------------------------------------------------------------------------
__global__ __launch_bounds__(256)
void ln_kernel(const float* __restrict__ x, const float* __restrict__ att,
               const float* __restrict__ gamma, const float* __restrict__ beta,
               float* __restrict__ out) {
    __shared__ float r1[256];
    __shared__ float r2[256];
    const int b = blockIdx.x;
    const int t = threadIdx.x;
    float y[3];
    float s = 0.f, sq = 0.f;
    for (int i = 0; i < 3; ++i) {
        const int d = t + i * 256;
        y[i] = x[(size_t)b * DD + d] + att[(size_t)b * DD + d];
        s += y[i];
        sq += y[i] * y[i];
    }
    r1[t] = s; r2[t] = sq;
    __syncthreads();
    for (int stp = 128; stp > 0; stp >>= 1) {
        if (t < stp) { r1[t] += r1[t + stp]; r2[t] += r2[t + stp]; }
        __syncthreads();
    }
    const float mu  = r1[0] * (1.0f / DD);
    const float var = r2[0] * (1.0f / DD) - mu * mu;
    const float inv = rsqrtf(var + 1e-5f);
    for (int i = 0; i < 3; ++i) {
        const int d = t + i * 256;
        out[(size_t)b * DD + d] = (y[i] - mu) * inv * gamma[d] + beta[d];
    }
}

// ---------------------------------------------------------------------------
static inline size_t alignup(size_t v) { return (v + 255) & ~(size_t)255; }

extern "C" void kernel_launch(void* const* d_in, const int* in_sizes, int n_in,
                              void* d_out, int out_size, void* d_ws, size_t ws_size,
                              hipStream_t stream) {
    const float* x      = (const float*)d_in[0];   // B x D
    const float* keys   = (const float*)d_in[1];   // P x D
    const float* values = (const float*)d_in[2];   // P x L x D
    const float* ipw    = (const float*)d_in[3];   // 3D x D
    const float* ipb    = (const float*)d_in[4];   // 3D
    const float* outw   = (const float*)d_in[5];   // D x D
    const float* outb   = (const float*)d_in[6];   // D
    const float* gamma  = (const float*)d_in[7];   // D
    const float* beta   = (const float*)d_in[8];   // D
    float* out = (float*)d_out;

    char* ws = (char*)d_ws;
    size_t off = 0;
    auto take = [&](size_t bytes) { char* p = ws + off; off += alignup(bytes); return p; };

    float*    knT   = (float*)   take((size_t)DD * PP * 4);
    _Float16* xh    = (_Float16*)take((size_t)BB * DD * 2);
    _Float16* wh    = (_Float16*)take((size_t)3 * DD * DD * 2);
    _Float16* owh   = (_Float16*)take((size_t)DD * DD * 2);
    _Float16* vh    = (_Float16*)take((size_t)PL * DD * 2);
    float*    kproj = (float*)   take((size_t)PL * DD * 4);
    float*    vproj = (float*)   take((size_t)PL * DD * 4);
    float*    qbuf  = (float*)   take((size_t)BB * DD * 4);
    int*      tidx  = (int*)     take((size_t)BB * KK * 4);
    _Float16* ctxh  = (_Float16*)take((size_t)BB * DD * 2);
    float*    attd  = (float*)   take((size_t)BB * DD * 4);
    (void)ws_size;

    // 1) f16 conversions
    cvt_f32_to_f16<<<1024, 256, 0, stream>>>(x,      xh,  BB * DD);
    cvt_f32_to_f16<<<1024, 256, 0, stream>>>(ipw,    wh,  3 * DD * DD);
    cvt_f32_to_f16<<<1024, 256, 0, stream>>>(outw,   owh, DD * DD);
    cvt_f32_to_f16<<<1024, 256, 0, stream>>>(values, vh,  PL * DD);

    // 2) normalized keys (transposed)
    keynorm_kernel<<<PP, 128, 0, stream>>>(keys, knT);

    // 3) project the 800 distinct value rows with wk / wv (WMMA + TDM)
    {
        dim3 g((PL + 127) / 128, DD / 128);
        gemm_f16_wmma<<<g, 256, 0, stream>>>(vh, wh + (size_t)DD * DD,
                                             ipb + DD, kproj, PL, DD, DD);
        gemm_f16_wmma<<<g, 256, 0, stream>>>(vh, wh + (size_t)2 * DD * DD,
                                             ipb + 2 * DD, vproj, PL, DD, DD);
    }

    // 4) q = x @ wq^T + bq
    {
        dim3 g(BB / 128, DD / 128);
        gemm_f16_wmma<<<g, 256, 0, stream>>>(xh, wh, ipb, qbuf, BB, DD, DD);
    }

    // 5) top-5 prompt selection per row
    topk_kernel<<<BB, 128, 0, stream>>>(x, knT, tidx);

    // 6) attention over gathered projected rows -> ctx (f16)
    attn_kernel<<<BB, 256, 0, stream>>>(qbuf, kproj, vproj, tidx, ctxh);

    // 7) attended = ctx @ out_w^T + out_b
    {
        dim3 g(BB / 128, DD / 128);
        gemm_f16_wmma<<<g, 256, 0, stream>>>(ctxh, owh, outb, attd, BB, DD, DD);
    }

    // 8) residual + layernorm
    ln_kernel<<<BB, 256, 0, stream>>>(x, attd, gamma, beta, out);
}
